// Linear_89300960019074
// MI455X (gfx1250) — compile-verified, matrix operand order checked by reading, not run on hardware
//
#include <hip/hip_runtime.h>
#include <stdint.h>

typedef __attribute__((ext_vector_type(16))) _Float16 v16h;
typedef __attribute__((ext_vector_type(8)))  float    v8f;
typedef __attribute__((ext_vector_type(2)))  _Float16 v2h;
typedef __attribute__((ext_vector_type(4)))  int      v4i;

#define BM 128
#define BN 128
#define BK 64
#define LDT 72                 // tile pitch in halves (BK + 8 pad)
#define TILE (BM * LDT)

#if __has_builtin(__builtin_amdgcn_global_load_async_to_lds_b128) && \
    __has_builtin(__builtin_amdgcn_s_wait_asynccnt)
#define USE_ASYNC 1
#else
#define USE_ASYNC 0
#endif

#if USE_ASYNC
// Per-lane 16B copy global -> LDS, tracked by ASYNCcnt (CDNA5 async path).
// Builtin signature (probe-verified): (v4i AS1*, v4i AS3*, Imm, Imm).
__device__ __forceinline__ void cp_async_b128(const void* g, void* l) {
    __builtin_amdgcn_global_load_async_to_lds_b128(
        (__attribute__((address_space(1))) v4i*)(uintptr_t)g,
        (__attribute__((address_space(3))) v4i*)(uint32_t)(uintptr_t)l,
        0, 0);
}
#endif

// 16x(K=32) 16-bit WMMA operand fragment from an LDS tile with K contiguous.
// lanes 0-15: rows, kb=0; lanes 16-31: same rows, kb=8; VGPR0-3 = K kb..kb+7,
// VGPR4-7 = K kb+16..kb+23  -> two contiguous ds_load_b128 per lane.
__device__ __forceinline__ v16h ldsm_frag(const _Float16* __restrict__ base,
                                          int row0, int pitch, int lane) {
    const int r  = row0 + (lane & 15);
    const int kb = (lane >> 4) << 3;
    union { int4 q[2]; v16h h; } u;
    const _Float16* p = base + r * pitch + kb;
    u.q[0] = *(const int4*)(p);
    u.q[1] = *(const int4*)(p + 16);
    return u.h;
}

// Magic-number dequant: 0x6400|w == fp16(1024+w); (1024+w)-(1024+z) is exact
// in packed fp16, then one v_pk_mul_f16 by the splatted scale.
// q[0..3] are 4 adjacent k-rows of one int32 column -> 8x ds_store_b64.
__device__ __forceinline__ void dequant_store(_Float16* __restrict__ Bt,
                                              const int* q,
                                              const v2h* zpk, const v2h* spk,
                                              int c16, int kk) {
    #pragma unroll
    for (int j = 0; j < 8; ++j) {
        union { int i[2]; v2h h[2]; } w;
        w.i[0] = 0x64006400 | ((q[0] >> (4 * j)) & 0xF)
                            | (((q[1] >> (4 * j)) & 0xF) << 16);
        w.i[1] = 0x64006400 | ((q[2] >> (4 * j)) & 0xF)
                            | (((q[3] >> (4 * j)) & 0xF) << 16);
        union { v2h h[2]; int2 p; } r;
        r.h[0] = (w.h[0] - zpk[j]) * spk[j];
        r.h[1] = (w.h[1] - zpk[j]) * spk[j];
        *(int2*)(Bt + ((c16 << 3) + j) * LDT + kk) = r.p;   // k..k+3, 8B aligned
    }
}

// Per-group (GROUP=128) metadata -> packed registers.
__device__ __forceinline__ void load_group_meta(int zq, const _Float16* scp,
                                                v2h* zpk, v2h* spk) {
    union { int4 v; _Float16 h[8]; } su;
    su.v = *(const int4*)scp;
    #pragma unroll
    for (int j = 0; j < 8; ++j) {
        const int z = (zq >> (4 * j)) & 0xF;
        union { int i; v2h h; } zz;
        zz.i = 0x64006400 | z | (z << 16);
        zpk[j] = zz.h;
        v2h sc; sc[0] = su.h[j]; sc[1] = su.h[j];
        spk[j] = sc;
    }
}

__global__ __launch_bounds__(256)
void w4a16_wmma_gemm(const _Float16* __restrict__ x,
                     const int*      __restrict__ qweight,
                     const int*      __restrict__ qzeros,
                     const _Float16* __restrict__ scales,
                     _Float16*       __restrict__ out,
                     int M, int N, int K)
{
    __shared__ __align__(16) _Float16 As[2][TILE];   // [m][k], double buffered
    __shared__ __align__(16) _Float16 Bs[2][TILE];   // [n][k], double buffered

    const int tid  = threadIdx.x;
    const int lane = tid & 31;
    const int wave = tid >> 5;
    const int wm   = wave >> 2;            // 0..1 -> 64-row band
    const int wn   = wave & 3;             // 0..3 -> 32-col band

    const int block_m = blockIdx.y * BM;
    const int block_n = blockIdx.x * BN;
    const int Nq      = N >> 3;

    // B dequant: int32 column c16 at 4 adjacent k rows (kk..kk+3)
    const int c16 = tid & 15;
    const int kk  = (tid >> 4) << 2;       // 0..60, multiple of 4
    const int qc  = (block_n >> 3) + c16;

    // A staging: 1024 16B chunks (128 rows x 8), 4 per thread
    const int a_row = tid >> 3;            // + 32*i, i=0..3
    const int a_off = (tid & 7) << 3;      // halves

    const _Float16* xp  = x + (size_t)(block_m + a_row) * K + a_off;
    const int*      qwp = qweight + (size_t)kk * Nq + qc;
    const int*      qzp = qzeros + qc;
    const _Float16* scp = scales + block_n + (c16 << 3);

    v8f acc[4][2] = {};
    v2h zpk[8], spk[8];
    int q[4];

    // ---- prologue: group 0 metadata + stage tile 0 into buffer 0 ----
    load_group_meta(qzp[0], scp, zpk, spk);
    #pragma unroll
    for (int r = 0; r < 4; ++r) q[r] = qwp[(size_t)r * Nq];
    qwp += (size_t)BK * Nq;

    #pragma unroll
    for (int i = 0; i < 4; ++i) {
#if USE_ASYNC
        cp_async_b128(xp + (size_t)(32 * i) * K,
                      &As[0][(a_row + 32 * i) * LDT + a_off]);
#else
        *(int4*)&As[0][(a_row + 32 * i) * LDT + a_off] =
            *(const int4*)(xp + (size_t)(32 * i) * K);
#endif
    }
    xp += BK;

    dequant_store(&Bs[0][0], q, zpk, spk, c16, kk);

    // ---- main pipelined loop (BK=64: 16 WMMAs per barrier) ----
    const int KT = K / BK;
    for (int kt = 0; kt < KT; ++kt) {
        const int  cur      = kt & 1;
        const int  nxt      = cur ^ 1;
        const bool has_next = (kt + 1) < KT;

        int4 ar[4] = {};
        if (has_next) {
            // next tile's weights/metadata: issued pre-barrier, used post-WMMA
            #pragma unroll
            for (int r = 0; r < 4; ++r) q[r] = qwp[(size_t)r * Nq];
            qwp += (size_t)BK * Nq;
            if (((kt + 1) & 1) == 0) {          // group boundary every 2 tiles
                qzp += Nq; scp += N;
                load_group_meta(qzp[0], scp, zpk, spk);
            }
#if !USE_ASYNC
            #pragma unroll
            for (int i = 0; i < 4; ++i)
                ar[i] = *(const int4*)(xp + (size_t)(32 * i) * K);
            xp += BK;
#endif
            __builtin_prefetch(qwp, 0, 0);      // global_prefetch_b8
        }

#if USE_ASYNC
        __builtin_amdgcn_s_wait_asynccnt(0);    // our As[cur] chunks landed
#endif
        __syncthreads();                        // tile[cur] ready everywhere;
                                                // all reads of [nxt] retired
        if (has_next) {
#if USE_ASYNC
            #pragma unroll
            for (int i = 0; i < 4; ++i)
                cp_async_b128(xp + (size_t)(32 * i) * K,
                              &As[nxt][(a_row + 32 * i) * LDT + a_off]);
            xp += BK;
#else
            #pragma unroll
            for (int i = 0; i < 4; ++i)
                *(int4*)&As[nxt][(a_row + 32 * i) * LDT + a_off] = ar[i];
#endif
        }

        // ---- two K=32 steps: 16 WMMAs on buffer[cur] ----
        #pragma unroll
        for (int ks = 0; ks < 2; ++ks) {
            const int ko = ks * 32;             // half offset within a row
            v16h afrag[4];
            #pragma unroll
            for (int i = 0; i < 4; ++i)
                afrag[i] = ldsm_frag(&As[cur][ko], wm * 64 + i * 16, LDT, lane);
            v16h bfrag[2];
            #pragma unroll
            for (int j = 0; j < 2; ++j)
                bfrag[j] = ldsm_frag(&Bs[cur][ko], wn * 32 + j * 16, LDT, lane);

            #pragma unroll
            for (int j = 0; j < 2; ++j) {
                #pragma unroll
                for (int i = 0; i < 4; ++i) {
                    acc[i][j] = __builtin_amdgcn_wmma_f32_16x16x32_f16(
                        false, afrag[i], false, bfrag[j],
                        (short)0, acc[i][j], false, false);
                }
            }
        }

        // ---- dequant tile kt+1 into Bs[nxt] (regs loaded pre-barrier) ----
        if (has_next)
            dequant_store(&Bs[nxt][0], q, zpk, spk, c16, kk);
    }

    // ---- epilogue: C/D layout (VGPR r -> M=r | r+8; lane[3:0] -> N) ----
    #pragma unroll
    for (int i = 0; i < 4; ++i) {
        #pragma unroll
        for (int j = 0; j < 2; ++j) {
            const int row0 = block_m + wm * 64 + i * 16 + ((lane >> 4) << 3);
            const int col  = block_n + wn * 32 + j * 16 + (lane & 15);
            #pragma unroll
            for (int r = 0; r < 8; ++r)
                out[(size_t)(row0 + r) * N + col] = (_Float16)acc[i][j][r];
        }
    }
}

extern "C" void kernel_launch(void* const* d_in, const int* in_sizes, int n_in,
                              void* d_out, int out_size, void* d_ws, size_t ws_size,
                              hipStream_t stream) {
    const _Float16* x  = (const _Float16*)d_in[0];
    const int*      qw = (const int*)d_in[1];
    const int*      qz = (const int*)d_in[2];
    const _Float16* sc = (const _Float16*)d_in[3];
    _Float16*       o  = (_Float16*)d_out;

    // Ratios are degenerate (GROUP_SIZE fixes them), so pin K/N and derive M.
    const int K = 4096;
    const int N = 11008;
    const int M = in_sizes[0] / K;   // 8192

    dim3 grid(N / BN, M / BM);       // 86 x 64 blocks
    w4a16_wmma_gemm<<<grid, 256, 0, stream>>>(x, qw, qz, sc, o, M, N, K);
}
